// LSReorderingLayer_46591805227609
// MI455X (gfx1250) — matrix operand order repaired and checked by reading, not compiled, not run on hardware
//
#include <hip/hip_runtime.h>
#include <hip/hip_bf16.h>
#include <math.h>

// -----------------------------------------------------------------------------
// Problem: S=2048, B=8, D=2048  ->  M = S*B = 16384 rows, K = N = D = 2048.
//   pp   = tanh(layer_input @ W1^T + x @ W2^T + b1 + b2)
//   gate = sigmoid(pp @ V^T + bV)
//   out  = gate * position_embedding
// ~412 GFLOP vs ~0.54 GB HBM traffic -> matrix-core bound: bf16 WMMA
// (v_wmma_f32_16x16x32_bf16, fp32 accum). bf16 weights (8MB each) live in the
// 192MB L2, so register-blocked direct global loads hit L2 with high reuse.
//
// This revision steers the K-loop loads into GVS addressing (uniform SGPR64
// base advanced by s_add each iteration + loop-invariant per-lane i32 byte
// offsets) to eliminate the per-iteration v_add_nc_u64 address VALU ops seen
// in the previous disassembly.
// -----------------------------------------------------------------------------

typedef __attribute__((ext_vector_type(8)))  float  v8f;
typedef __attribute__((ext_vector_type(16))) __bf16 v16bf;
typedef __attribute__((ext_vector_type(8)))  __bf16 v8bf;

union FragU { v16bf v; struct { v8bf lo, hi; } p; };

static constexpr int S_ = 2048;
static constexpr int B_ = 8;
static constexpr int D_ = 2048;            // K and N
static constexpr int M_ = S_ * B_;         // 16384

// round-to-nearest-even f32 -> bf16
static __device__ __forceinline__ unsigned short f2bf(float x) {
    union { float f; unsigned u; } c; c.f = x;
    unsigned r = c.u + 0x7FFFu + ((c.u >> 16) & 1u);
    return (unsigned short)(r >> 16);
}

// ---------------------------- fp32 -> bf16 convert ---------------------------
__global__ __launch_bounds__(256) void f32_to_bf16_kernel(
    const float* __restrict__ in, unsigned short* __restrict__ out, int n)
{
    int i = (blockIdx.x * 256 + threadIdx.x) * 4;
    if (i >= n) return;
    const float4 v = *(const float4*)(in + i);
    ushort4 o;
    o.x = f2bf(v.x); o.y = f2bf(v.y); o.z = f2bf(v.z); o.w = f2bf(v.w);
    *(ushort4*)(out + i) = o;
}

// ------------------------ shared WMMA K-accumulation -------------------------
// Wave computes a 64x64 tile: acc[i][j] is the 16x16 fragment at
// rows m0+16*i.., cols n0+16*j..   (lm = lane%16, lh = lane/16)
//
// A frag layout (16-bit A 16x32): lane lm (+16*lh) holds row m0+16*i+lm,
//   K elements [lh*8 .. lh*8+7] in v0..3 and [16+lh*8 .. +7] in v4..7.
// B frag layout (16-bit B 32x16): lane holds col n0+16*j+lm,
//   K elements [lh*8 .. lh*8+7] in v0..3 and [16+lh*8 .. +7] in v4..7.
static __device__ __forceinline__ void wmma_k_pass(
    const __bf16* __restrict__ A,   // M_ x D_, row-major, K contiguous
    const __bf16* __restrict__ W,   // N  x D_, row-major, K contiguous (B = W^T)
    int m0, int n0, int lm, int lh, v8f acc[4][4])
{
    // loop-invariant per-lane element offsets (fit in i32: < 2^25 elements)
    int aoff[4], boff[4];
#pragma unroll
    for (int i = 0; i < 4; ++i)
        aoff[i] = (m0 + 16 * i + lm) * D_ + lh * 8;
#pragma unroll
    for (int j = 0; j < 4; ++j)
        boff[j] = (n0 + 16 * j + lm) * D_ + lh * 8;

    // uniform bases advanced by scalar adds (SGPR) each iteration
    const __bf16* Ak = A;
    const __bf16* Wk = W;

    for (int k0 = 0; k0 < D_; k0 += 32) {
        v16bf a[4], b[4];
#pragma unroll
        for (int i = 0; i < 4; ++i) {
            FragU f;
            f.p.lo = *(const v8bf*)(Ak + aoff[i]);
            f.p.hi = *(const v8bf*)(Ak + aoff[i] + 16);
            a[i] = f.v;
        }
#pragma unroll
        for (int j = 0; j < 4; ++j) {
            FragU f;
            f.p.lo = *(const v8bf*)(Wk + boff[j]);
            f.p.hi = *(const v8bf*)(Wk + boff[j] + 16);
            b[j] = f.v;
        }
#pragma unroll
        for (int i = 0; i < 4; ++i)
#pragma unroll
            for (int j = 0; j < 4; ++j)
                acc[i][j] = __builtin_amdgcn_wmma_f32_16x16x32_bf16(
                    false, a[i], false, b[j], (short)0, acc[i][j], false, false);
        Ak += 32;
        Wk += 32;
    }
}

// --------------------------- layer 1: fused dual GEMM ------------------------
// H = tanh(A1 @ W1^T + A2 @ W2^T + b1 + b2), bf16 output
__global__ __launch_bounds__(256) void gemm_tanh_kernel(
    const __bf16* __restrict__ A1, const __bf16* __restrict__ A2,
    const __bf16* __restrict__ W1, const __bf16* __restrict__ W2,
    const float* __restrict__ b1, const float* __restrict__ b2,
    unsigned short* __restrict__ H)
{
    const int tid  = threadIdx.x;
    const int wave = tid >> 5;
    const int lane = tid & 31;
    const int lm = lane & 15, lh = lane >> 4;
    const int m0 = blockIdx.x * 128 + (wave & 1) * 64;   // 128 blocks along M
    const int n0 = blockIdx.y * 256 + (wave >> 1) * 64;  // 8 blocks along N

    v8f acc[4][4] = {};
    wmma_k_pass(A1, W1, m0, n0, lm, lh, acc);   // += layer_input @ W1^T
    wmma_k_pass(A2, W2, m0, n0, lm, lh, acc);   // += x @ W2^T

    // epilogue: C/D layout -> row = 16*i + r + 8*lh, col = 16*j + lm
#pragma unroll
    for (int j = 0; j < 4; ++j) {
        const int col = n0 + 16 * j + lm;
        const float bias = b1[col] + b2[col];
#pragma unroll
        for (int i = 0; i < 4; ++i) {
#pragma unroll
            for (int r = 0; r < 8; ++r) {
                const int row = m0 + 16 * i + r + lh * 8;
                H[(size_t)row * D_ + col] = f2bf(tanhf(acc[i][j][r] + bias));
            }
        }
    }
}

// ------------------ layer 2: GEMM + sigmoid gate * pos_embed -----------------
__global__ __launch_bounds__(256) void gemm_gate_kernel(
    const __bf16* __restrict__ H, const __bf16* __restrict__ Vw,
    const float* __restrict__ bV, const float* __restrict__ pe,
    float* __restrict__ out)
{
    const int tid  = threadIdx.x;
    const int wave = tid >> 5;
    const int lane = tid & 31;
    const int lm = lane & 15, lh = lane >> 4;
    const int m0 = blockIdx.x * 128 + (wave & 1) * 64;
    const int n0 = blockIdx.y * 256 + (wave >> 1) * 64;

    v8f acc[4][4] = {};
    wmma_k_pass(H, Vw, m0, n0, lm, lh, acc);

#pragma unroll
    for (int j = 0; j < 4; ++j) {
        const int col = n0 + 16 * j + lm;
        const float bias = bV[col];
#pragma unroll
        for (int i = 0; i < 4; ++i) {
#pragma unroll
            for (int r = 0; r < 8; ++r) {
                const int row = m0 + 16 * i + r + lh * 8;
                const size_t idx = (size_t)row * D_ + col;
                const float g = 1.0f / (1.0f + expf(-(acc[i][j][r] + bias)));
                out[idx] = g * pe[idx];
            }
        }
    }
}

// --------------------------------- launcher ----------------------------------
extern "C" void kernel_launch(void* const* d_in, const int* in_sizes, int n_in,
                              void* d_out, int out_size, void* d_ws, size_t ws_size,
                              hipStream_t stream)
{
    // inputs (setup_inputs order):
    const float* x   = (const float*)d_in[0];   // (S,B,D)
    const float* li  = (const float*)d_in[1];   // layer_input (S,B,D)
    const float* pe  = (const float*)d_in[2];   // position_embedding (S,B,D)
    // d_in[3] encoder_padding_mask, d_in[4] langs: unused (uniform path)
    const float* W1  = (const float*)d_in[5];
    const float* b1  = (const float*)d_in[6];
    const float* W2  = (const float*)d_in[7];
    const float* b2  = (const float*)d_in[8];
    const float* Vw  = (const float*)d_in[9];
    const float* bV  = (const float*)d_in[10];
    float* out = (float*)d_out;

    // workspace layout (bytes)
    char* ws = (char*)d_ws;
    const size_t actB = (size_t)M_ * D_ * 2;   // 64 MiB  (bf16 activations)
    const size_t wgtB = (size_t)D_ * D_ * 2;   //  8 MiB  (bf16 weights)
    unsigned short* A1b = (unsigned short*)(ws);                          // layer_input bf16
    unsigned short* A2b = (unsigned short*)(ws + actB);                   // x bf16
    unsigned short* W1b = (unsigned short*)(ws + 2 * actB);
    unsigned short* W2b = (unsigned short*)(ws + 2 * actB + wgtB);
    unsigned short* Vb  = (unsigned short*)(ws + 2 * actB + 2 * wgtB);
    unsigned short* Hb  = (unsigned short*)(ws + 2 * actB + 3 * wgtB);    // pp bf16

    const int nAct = M_ * D_;        // 33,554,432
    const int nWgt = D_ * D_;        //  4,194,304
    dim3 blk(256);
    f32_to_bf16_kernel<<<dim3(nAct / 4 / 256), blk, 0, stream>>>(li, A1b, nAct);
    f32_to_bf16_kernel<<<dim3(nAct / 4 / 256), blk, 0, stream>>>(x,  A2b, nAct);
    f32_to_bf16_kernel<<<dim3(nWgt / 4 / 256), blk, 0, stream>>>(W1, W1b, nWgt);
    f32_to_bf16_kernel<<<dim3(nWgt / 4 / 256), blk, 0, stream>>>(W2, W2b, nWgt);
    f32_to_bf16_kernel<<<dim3(nWgt / 4 / 256), blk, 0, stream>>>(Vw, Vb,  nWgt);

    dim3 grid(M_ / 128, D_ / 256);   // 128 x 8 workgroups, 8 waves each
    gemm_tanh_kernel<<<grid, blk, 0, stream>>>(
        (const __bf16*)A1b, (const __bf16*)A2b,
        (const __bf16*)W1b, (const __bf16*)W2b, b1, b2, Hb);
    gemm_gate_kernel<<<grid, blk, 0, stream>>>(
        (const __bf16*)Hb, (const __bf16*)Vb, bV, pe, out);
}
